// DGCNN_DA_28862180229721
// MI455X (gfx1250) — matrix operand order repaired and checked by reading, not compiled, not run on hardware
//
#include <hip/hip_runtime.h>
#include <hip/hip_fp16.h>

typedef __attribute__((ext_vector_type(16))) _Float16 v16h;
typedef __attribute__((ext_vector_type(8)))  _Float16 v8h;
typedef __attribute__((ext_vector_type(8)))  float    v8f;

#define BNEPS 1e-5f
#define SLOPE 0.2f
#define NPTS 1024
#define NB 32
#define KNN 20
#define CCAT 640
#define NEG_BIG -3.4e38f

__device__ __forceinline__ float leaky(float y){ return y > 0.f ? y : SLOPE * y; }

__device__ __forceinline__ v8f wmma_f16(v16h a, v16h b, v8f c){
  return __builtin_amdgcn_wmma_f32_16x16x32_f16(false, a, false, b, (short)0, c, false, false);
}

__device__ __forceinline__ v16h cat16(v8h lo, v8h hi){
  return __builtin_shufflevector(lo, hi, 0,1,2,3,4,5,6,7,8,9,10,11,12,13,14,15);
}

// A operand tile 16x32 from row-major A[m][k] (lda elems): two contiguous b128 loads per lane.
__device__ __forceinline__ v16h load_opA(const _Float16* __restrict__ A, int lda, int lane){
  int m = lane & 15, kh = lane >> 4;
  const _Float16* p = A + (size_t)m * lda + 16*kh;
  v8h lo = *(const v8h*)p;
  v8h hi = *(const v8h*)(p + 8);
  return cat16(lo, hi);
}

// B operand tile 32x16 where B[k][n] = Xt[n][k] (features N-major): two b128 loads per lane.
__device__ __forceinline__ v16h load_opB(const _Float16* __restrict__ Xt, int ldc,
                                         int n0, int k0, int lane){
  const _Float16* p = Xt + (size_t)(n0 + (lane & 15)) * ldc + k0 + 16*(lane >> 4);
  v8h lo = *(const v8h*)p;
  v8h hi = *(const v8h*)(p + 8);
  return cat16(lo, hi);
}

// Same pattern out of LDS (ds_load_b128 x2)
__device__ __forceinline__ v16h load_opB_lds(const _Float16* Xt, int ldc, int k0, int lane){
  const _Float16* p = Xt + (size_t)(lane & 15) * ldc + k0 + 16*(lane >> 4);
  v8h lo = *(const v8h*)p;
  v8h hi = *(const v8h*)(p + 8);
  return cat16(lo, hi);
}

// ---------------- stage-1 distance (C=3, VALU) ----------------
__global__ void dist3_kernel(const float* __restrict__ x, float* __restrict__ dist){
  const int N = NPTS;
  int b = blockIdx.z, m = blockIdx.y;
  int n = blockIdx.x * blockDim.x + threadIdx.x;
  const float* xb = x + (size_t)b * 3 * N;
  float d0 = xb[m]       - xb[n];
  float d1 = xb[N + m]   - xb[N + n];
  float d2 = xb[2*N + m] - xb[2*N + n];
  dist[((size_t)b * N + m) * N + n] = -(d0*d0 + d1*d1 + d2*d2);
}

// ---------------- top-20 per row: one wave32 per row ----------------
__global__ void topk_kernel(const float* __restrict__ dist, int* __restrict__ idx){
  const int N = NPTS;
  int row  = blockIdx.x * (blockDim.x >> 5) + (threadIdx.x >> 5);
  int lane = threadIdx.x & 31;
  const float* d = dist + (size_t)row * N;
  float vals[32];
#pragma unroll
  for (int i = 0; i < 32; i++) vals[i] = d[lane + 32*i];
  int* out = idx + (size_t)row * KNN;
  for (int k = 0; k < KNN; k++){
    float best = NEG_BIG; int bi = 0;
#pragma unroll
    for (int i = 0; i < 32; i++)
      if (vals[i] > best){ best = vals[i]; bi = lane + 32*i; }
#pragma unroll
    for (int sh = 16; sh; sh >>= 1){
      float ov = __shfl_xor(best, sh, 32);
      int   oi = __shfl_xor(bi,   sh, 32);
      if (ov > best || (ov == best && oi < bi)){ best = ov; bi = oi; }
    }
    if (lane == 0) out[k] = bi;
    if ((bi & 31) == lane) vals[bi >> 5] = NEG_BIG;
  }
}

// ------------- knn Gram distances via WMMA; C fixed at 128 (nk=4 static) -------------
__global__ void knn_dist_wmma_kernel(const _Float16* __restrict__ cat, int off,
                                     const float* __restrict__ xx, float* __restrict__ dist){
  const int N = NPTS;
  constexpr int C = 128;
  constexpr int NK = C / 32;                               // 4, compile-time
  int b = blockIdx.y, mt = blockIdx.x;
  int wave = threadIdx.x >> 5, lane = threadIdx.x & 31;
  const _Float16* Xt = cat + (size_t)b * N * CCAT + off;   // row n at Xt + n*640
  const float* xxb = xx + (size_t)b * N;
  float* D = dist + (size_t)b * N * N;
  int m0 = mt * 16;
  v16h a0 = load_opA(Xt + (size_t)m0 * CCAT + 0,  CCAT, lane);
  v16h a1 = load_opA(Xt + (size_t)m0 * CCAT + 32, CCAT, lane);
  v16h a2 = load_opA(Xt + (size_t)m0 * CCAT + 64, CCAT, lane);
  v16h a3 = load_opA(Xt + (size_t)m0 * CCAT + 96, CCAT, lane);
  (void)NK;
  for (int nt = wave; nt < 64; nt += 4){
    v8f acc = {};
    acc = wmma_f16(a0, load_opB(Xt, CCAT, nt*16, 0,  lane), acc);
    acc = wmma_f16(a1, load_opB(Xt, CCAT, nt*16, 32, lane), acc);
    acc = wmma_f16(a2, load_opB(Xt, CCAT, nt*16, 64, lane), acc);
    acc = wmma_f16(a3, load_opB(Xt, CCAT, nt*16, 96, lane), acc);
#pragma unroll
    for (int r = 0; r < 8; r++){
      int m = m0 + r + ((lane >= 16) ? 8 : 0);
      int n = nt*16 + (lane & 15);
      D[(size_t)m * N + n] = 2.f * acc[r] - xxb[m] - xxb[n];
    }
  }
}

// ------------- 1x1 conv + BN + LeakyReLU via WMMA, C compile-time -------------
template <int C>
__global__ void conv_bn_wmma_kernel(const _Float16* __restrict__ W,
                                    const _Float16* __restrict__ cat, int off,
                                    const float* __restrict__ s, const float* __restrict__ t,
                                    float* __restrict__ out, int O){
  const int N = NPTS;
  int b = blockIdx.z, ot = blockIdx.y;
  int wave = threadIdx.x >> 5, lane = threadIdx.x & 31;
  int nt = blockIdx.x * 4 + wave;
  const _Float16* Xt = cat + (size_t)b * N * CCAT + off;
  v8f acc = {};
#pragma unroll
  for (int kt = 0; kt < C/32; kt++){
    v16h a  = load_opA(W + (size_t)(ot*16) * C + kt*32, C, lane);
    v16h bb = load_opB(Xt, CCAT, nt*16, kt*32, lane);
    acc = wmma_f16(a, bb, acc);
  }
#pragma unroll
  for (int r = 0; r < 8; r++){
    int ch = ot*16 + r + ((lane >= 16) ? 8 : 0);
    int n  = nt*16 + (lane & 15);
    out[((size_t)b * O + ch) * N + n] = leaky(acc[r] * s[ch] + t[ch]);
  }
}

// ------- conv5 (1024x640) + BN + LeakyReLU + channel-max; async global->LDS tile staging -------
__global__ void conv5_max_kernel(const _Float16* __restrict__ W,
                                 const _Float16* __restrict__ cat,
                                 const float* __restrict__ s, const float* __restrict__ t,
                                 float* __restrict__ pooled){
  __shared__ _Float16 sX[16 * CCAT];     // 16 points x 640 channels, N-major
  __shared__ float sRed[4][16];
  const int N = NPTS;
  int b = blockIdx.y, n0 = blockIdx.x * 16;
  int tid = threadIdx.x, wave = tid >> 5, lane = tid & 31;
  const _Float16* Cb = cat + (size_t)b * N * CCAT;
  // async DMA: 1280 chunks of 8 f16 (16B) from global rows into LDS (ASYNCcnt-tracked)
  for (int c = tid; c < 1280; c += 128){
    int j = c / 80, w = c % 80;
    unsigned long long g = (unsigned long long)(Cb + (size_t)(n0 + j) * CCAT + w*8);
    unsigned l = (unsigned)(unsigned long long)(&sX[j * CCAT + w*8]);
    asm volatile("global_load_async_to_lds_b128 %0, %1, off" :: "v"(l), "v"(g) : "memory");
  }
  asm volatile("s_wait_asynccnt 0" ::: "memory");
  __syncthreads();
  float colmax = NEG_BIG;
  for (int ot = wave; ot < 64; ot += 4){
    v8f acc = {};
#pragma unroll
    for (int kt = 0; kt < 20; kt++){
      v16h a  = load_opA(W + (size_t)(ot*16) * CCAT + kt*32, CCAT, lane);
      v16h bb = load_opB_lds(sX, CCAT, kt*32, lane);
      acc = wmma_f16(a, bb, acc);
    }
#pragma unroll
    for (int r = 0; r < 8; r++){
      int ch = ot*16 + r + ((lane >= 16) ? 8 : 0);
      colmax = fmaxf(colmax, leaky(acc[r] * s[ch] + t[ch]));
    }
  }
  colmax = fmaxf(colmax, __shfl_xor(colmax, 16, 32));
  if (lane < 16) sRed[wave][lane] = colmax;
  __syncthreads();
  if (wave == 0 && lane < 16){
    float m = fmaxf(fmaxf(sRed[0][lane], sRed[1][lane]), fmaxf(sRed[2][lane], sRed[3][lane]));
    pooled[(size_t)b * N + n0 + lane] = m;
  }
}

// ---------------- small helper kernels ----------------
__global__ void conv1_kernel(const float* __restrict__ x, const float* __restrict__ w,
                             const float* __restrict__ s, const float* __restrict__ t,
                             float* __restrict__ out){
  const int N = NPTS;
  int gid = blockIdx.x * blockDim.x + threadIdx.x;
  int n = gid & (N-1);
  int o = (gid >> 10) & 63;
  int b = gid >> 16;
  const float* xb = x + (size_t)b * 3 * N;
  float y = w[o*3+0]*xb[n] + w[o*3+1]*xb[N+n] + w[o*3+2]*xb[2*N+n];
  out[((size_t)b * 64 + o) * N + n] = leaky(y * s[o] + t[o]);
}

__global__ void graph_feature_kernel(const float* __restrict__ h, const int* __restrict__ idx,
                                     float* __restrict__ out, int C){
  const int N = NPTS;
  int gid = blockIdx.x * blockDim.x + threadIdx.x;
  int n = gid & (N-1);
  int c = (gid >> 10) % C;
  int b = gid / (C * N);
  const float* hb = h + ((size_t)b * C + c) * N;
  const int* id = idx + ((size_t)b * N + n) * KNN;
  float center = hb[n];
  float mx = NEG_BIG;
#pragma unroll
  for (int k = 0; k < KNN; k++) mx = fmaxf(mx, hb[id[k]]);
  size_t base = (size_t)b * (2*C) * N;
  out[base + (size_t)c * N + n]       = mx - center;
  out[base + (size_t)(C + c) * N + n] = center;
}

__global__ void to_f16_kernel(const float* __restrict__ src, _Float16* __restrict__ dst, int count){
  int i = blockIdx.x * blockDim.x + threadIdx.x;
  if (i < count) dst[i] = (_Float16)src[i];
}

// (B,C,N) f32 -> N-major f16 slab cat[b][n][off+c]  (transpose during conversion)
__global__ void cat_f16_kernel(const float* __restrict__ src, _Float16* __restrict__ cat,
                               int off, int C){
  const int N = NPTS;
  int gid = blockIdx.x * blockDim.x + threadIdx.x;
  int n = gid & (N-1);
  int c = (gid >> 10) % C;
  int b = gid / (C * N);
  cat[((size_t)b * N + n) * CCAT + off + c] = (_Float16)src[((size_t)b * C + c) * N + n];
}

__global__ void colsq_kernel(const _Float16* __restrict__ cat, int off, int C,
                             float* __restrict__ xx){
  const int N = NPTS;
  int gid = blockIdx.x * blockDim.x + threadIdx.x;
  int n = gid & (N-1);
  int b = gid >> 10;
  const _Float16* row = cat + ((size_t)b * N + n) * CCAT + off;
  float acc = 0.f;
  for (int c = 0; c < C; c++){ float v = (float)row[c]; acc += v * v; }
  xx[(size_t)b * N + n] = acc;
}

__global__ void bn_prep_kernel(const float* g, const float* b, const float* m, const float* v,
                               float* s, float* t, int C){
  int i = blockIdx.x * blockDim.x + threadIdx.x;
  if (i < C){ float sc = g[i] * rsqrtf(v[i] + BNEPS); s[i] = sc; t[i] = b[i] - m[i] * sc; }
}

__global__ void fc_kernel(const float* __restrict__ in, const float* __restrict__ W,
                          const float* __restrict__ bias,
                          const float* __restrict__ g, const float* __restrict__ bb,
                          const float* __restrict__ mm, const float* __restrict__ vv,
                          float* __restrict__ out, int IN, int OUT, int act){
  int gid = blockIdx.x * blockDim.x + threadIdx.x;
  if (gid >= NB * OUT) return;
  int o = gid % OUT, r = gid / OUT;
  const float* ip = in + (size_t)r * IN;
  const float* wp = W + (size_t)o * IN;
  float acc = 0.f;
  for (int i = 0; i < IN; i++) acc += ip[i] * wp[i];
  if (bias) acc += bias[o];
  if (g){ float sc = g[o] * rsqrtf(vv[o] + BNEPS); acc = (acc - mm[o]) * sc + bb[o]; }
  if (act) acc = leaky(acc);
  out[gid] = acc;
}

extern "C" void kernel_launch(void* const* d_in, const int* in_sizes, int n_in,
                              void* d_out, int out_size, void* d_ws, size_t ws_size,
                              hipStream_t stream) {
  (void)in_sizes; (void)n_in; (void)out_size; (void)ws_size;
  const float* x   = (const float*)d_in[0];
  const float* w1  = (const float*)d_in[1];
  const float* g1  = (const float*)d_in[2];  const float* bb1 = (const float*)d_in[3];
  const float* m1  = (const float*)d_in[4];  const float* v1  = (const float*)d_in[5];
  const float* w2  = (const float*)d_in[6];
  const float* g2  = (const float*)d_in[7];  const float* bb2 = (const float*)d_in[8];
  const float* m2  = (const float*)d_in[9];  const float* v2  = (const float*)d_in[10];
  const float* w3  = (const float*)d_in[11];
  const float* g3  = (const float*)d_in[12]; const float* bb3 = (const float*)d_in[13];
  const float* m3  = (const float*)d_in[14]; const float* v3  = (const float*)d_in[15];
  const float* w4  = (const float*)d_in[16];
  const float* g4  = (const float*)d_in[17]; const float* bb4 = (const float*)d_in[18];
  const float* m4  = (const float*)d_in[19]; const float* v4  = (const float*)d_in[20];
  const float* w5  = (const float*)d_in[21];
  const float* g5  = (const float*)d_in[22]; const float* bb5 = (const float*)d_in[23];
  const float* m5  = (const float*)d_in[24]; const float* v5  = (const float*)d_in[25];
  const float* l1w = (const float*)d_in[26];
  const float* g6  = (const float*)d_in[27]; const float* bb6 = (const float*)d_in[28];
  const float* m6  = (const float*)d_in[29]; const float* v6  = (const float*)d_in[30];
  const float* l2w = (const float*)d_in[31]; const float* l2b = (const float*)d_in[32];
  const float* g7  = (const float*)d_in[33]; const float* bb7 = (const float*)d_in[34];
  const float* m7  = (const float*)d_in[35]; const float* v7  = (const float*)d_in[36];
  const float* l3w = (const float*)d_in[37]; const float* l3b = (const float*)d_in[38];
  float* out = (float*)d_out;

  char* wsb = (char*)d_ws;
  size_t off = 0;
  auto alloc = [&](size_t bytes) -> void* {
    void* p = wsb + off; off += (bytes + 255) & ~(size_t)255; return p;
  };
  float*     dist  = (float*)alloc((size_t)NB * NPTS * NPTS * 4);
  int*       idx   = (int*)  alloc((size_t)NB * NPTS * KNN * 4);
  float*     xx    = (float*)alloc((size_t)NB * NPTS * 4);
  float*     h64   = (float*)alloc((size_t)NB * 64  * NPTS * 4);
  float*     h128  = (float*)alloc((size_t)NB * 128 * NPTS * 4);
  float*     xbuf  = (float*)alloc((size_t)NB * 256 * NPTS * 4);
  _Float16*  cat   = (_Float16*)alloc((size_t)NB * NPTS * CCAT * 2); // N-major f16 features
  _Float16*  w2h   = (_Float16*)alloc((size_t)64 * 128 * 2);
  _Float16*  w3h   = (_Float16*)alloc((size_t)64 * 128 * 2);
  _Float16*  w4h   = (_Float16*)alloc((size_t)128 * 128 * 2);
  _Float16*  w5h   = (_Float16*)alloc((size_t)1024 * CCAT * 2);
  float*     s1 = (float*)alloc(64*4);    float* t1 = (float*)alloc(64*4);
  float*     s2 = (float*)alloc(64*4);    float* t2 = (float*)alloc(64*4);
  float*     s3 = (float*)alloc(64*4);    float* t3 = (float*)alloc(64*4);
  float*     s4 = (float*)alloc(128*4);   float* t4 = (float*)alloc(128*4);
  float*     s5 = (float*)alloc(1024*4);  float* t5 = (float*)alloc(1024*4);
  float*     pooled = (float*)alloc((size_t)NB * NPTS * 4);
  float*     h6  = (float*)alloc((size_t)NB * 512 * 4);
  float*     h7  = (float*)alloc((size_t)NB * 256 * 4);

  const int T = 256;
  to_f16_kernel<<<(64*128 + T-1)/T, T, 0, stream>>>(w2, w2h, 64*128);
  to_f16_kernel<<<(64*128 + T-1)/T, T, 0, stream>>>(w3, w3h, 64*128);
  to_f16_kernel<<<(128*128 + T-1)/T, T, 0, stream>>>(w4, w4h, 128*128);
  to_f16_kernel<<<(1024*CCAT + T-1)/T, T, 0, stream>>>(w5, w5h, 1024*CCAT);
  bn_prep_kernel<<<1, 64, 0, stream>>>(g1, bb1, m1, v1, s1, t1, 64);
  bn_prep_kernel<<<1, 64, 0, stream>>>(g2, bb2, m2, v2, s2, t2, 64);
  bn_prep_kernel<<<1, 64, 0, stream>>>(g3, bb3, m3, v3, s3, t3, 64);
  bn_prep_kernel<<<1, 128, 0, stream>>>(g4, bb4, m4, v4, s4, t4, 128);
  bn_prep_kernel<<<4, 256, 0, stream>>>(g5, bb5, m5, v5, s5, t5, 1024);

  // stage 1
  dist3_kernel<<<dim3(4, NPTS, NB), T, 0, stream>>>(x, dist);
  topk_kernel<<<(NB*NPTS)/8, T, 0, stream>>>(dist, idx);
  conv1_kernel<<<(NB*64*NPTS)/T, T, 0, stream>>>(x, w1, s1, t1, h64);
  graph_feature_kernel<<<(NB*64*NPTS)/T, T, 0, stream>>>(h64, idx, xbuf, 64);

  // stage 2
  cat_f16_kernel<<<(NB*128*NPTS)/T, T, 0, stream>>>(xbuf, cat, 0, 128);
  colsq_kernel<<<(NB*NPTS)/T, T, 0, stream>>>(cat, 0, 128, xx);
  knn_dist_wmma_kernel<<<dim3(64, NB), 128, 0, stream>>>(cat, 0, xx, dist);
  topk_kernel<<<(NB*NPTS)/8, T, 0, stream>>>(dist, idx);
  conv_bn_wmma_kernel<128><<<dim3(16, 4, NB), 128, 0, stream>>>(w2h, cat, 0, s2, t2, h64, 64);
  graph_feature_kernel<<<(NB*64*NPTS)/T, T, 0, stream>>>(h64, idx, xbuf, 64);

  // stage 3
  cat_f16_kernel<<<(NB*128*NPTS)/T, T, 0, stream>>>(xbuf, cat, 128, 128);
  colsq_kernel<<<(NB*NPTS)/T, T, 0, stream>>>(cat, 128, 128, xx);
  knn_dist_wmma_kernel<<<dim3(64, NB), 128, 0, stream>>>(cat, 128, xx, dist);
  topk_kernel<<<(NB*NPTS)/8, T, 0, stream>>>(dist, idx);
  conv_bn_wmma_kernel<128><<<dim3(16, 4, NB), 128, 0, stream>>>(w3h, cat, 128, s3, t3, h64, 64);
  graph_feature_kernel<<<(NB*64*NPTS)/T, T, 0, stream>>>(h64, idx, xbuf, 64);

  // stage 4
  cat_f16_kernel<<<(NB*128*NPTS)/T, T, 0, stream>>>(xbuf, cat, 256, 128);
  colsq_kernel<<<(NB*NPTS)/T, T, 0, stream>>>(cat, 256, 128, xx);
  knn_dist_wmma_kernel<<<dim3(64, NB), 128, 0, stream>>>(cat, 256, xx, dist);
  topk_kernel<<<(NB*NPTS)/8, T, 0, stream>>>(dist, idx);
  conv_bn_wmma_kernel<128><<<dim3(16, 8, NB), 128, 0, stream>>>(w4h, cat, 256, s4, t4, h128, 128);
  graph_feature_kernel<<<(NB*128*NPTS)/T, T, 0, stream>>>(h128, idx, xbuf, 128);
  cat_f16_kernel<<<(NB*256*NPTS)/T, T, 0, stream>>>(xbuf, cat, 384, 256);

  // conv5 + channel-max (fused)
  conv5_max_kernel<<<dim3(64, NB), 128, 0, stream>>>(w5h, cat, s5, t5, pooled);

  // FC head
  fc_kernel<<<(NB*512)/T, T, 0, stream>>>(pooled, l1w, nullptr, g6, bb6, m6, v6, h6, 1024, 512, 1);
  fc_kernel<<<(NB*256)/T, T, 0, stream>>>(h6, l2w, l2b, g7, bb7, m7, v7, h7, 512, 256, 1);
  fc_kernel<<<(NB*40 + T-1)/T, T, 0, stream>>>(h7, l3w, l3b, nullptr, nullptr, nullptr, nullptr,
                                               out, 256, 40, 0);
}
